// CausalGraphReasoning_74053826118333
// MI455X (gfx1250) — matrix operand order repaired and checked by reading, not compiled, not run on hardware
//
#include <hip/hip_runtime.h>

#define B_   256
#define N_   196
#define D_   768
#define H_   256
#define KK   8
#define MT   13            // ceil(N_/16)
#define NEGV (-1.0e9f)
#define PADV (-3.0e38f)

typedef __attribute__((ext_vector_type(16))) _Float16 v16h;
typedef __attribute__((ext_vector_type(8)))  _Float16 v8h;
typedef __attribute__((ext_vector_type(8)))  float    v8f;

__device__ __forceinline__ v8f wmma_f16(v16h a, v16h b, v8f c) {
  // D = A(16x32 f16) * B(32x16 f16) + C(16x16 f32)
  return __builtin_amdgcn_wmma_f32_16x16x32_f16(false, a, false, b, (short)0, c,
                                                false, false);
}

__device__ __forceinline__ v16h cat8(v8h lo, v8h hi) {
  union { v16h v; v8h h[2]; } u;
  u.h[0] = lo; u.h[1] = hi;
  return u.v;
}

// ---- fragment loaders (wave32; lane = hf*16 + l16) ----------------------
// 16-bit A layout (16x32): lane l16 holds row l16; hf=0 -> K {0..7,16..23},
// hf=1 -> K {8..15,24..31}.
__device__ __forceinline__ v16h load_a_f32(const float* __restrict__ A, int lda,
                                           int row, int k0, int hf) {
  v16h a;
  const float* p = A + (size_t)row * lda + k0 + hf * 8;
#pragma unroll
  for (int i = 0; i < 8; ++i) a[i] = (_Float16)p[i];
#pragma unroll
  for (int i = 0; i < 8; ++i) a[8 + i] = (_Float16)p[16 + i];
  return a;
}

__device__ __forceinline__ v16h load_a_f16(const _Float16* __restrict__ A, int lda,
                                           int row, int k0, int hf) {
  const _Float16* p = A + (size_t)row * lda + k0 + hf * 8;   // 16B aligned
  return cat8(*(const v8h*)p, *(const v8h*)(p + 16));
}

// B layout (32x16, K x N): lane l16 holds col l16; hf=0 -> K {0..7,16..23},
// hf=1 -> K {8..15,24..31}.  Source is K-contiguous per column (transposed weight
// WT[col][k], or node rows for n·nᵀ) -> two 16-byte chunks per lane.
__device__ __forceinline__ v16h load_bT_f16(const _Float16* __restrict__ WT, int ldT,
                                            int col, int k0, int hf) {
  const _Float16* p = WT + (size_t)col * ldT + k0 + hf * 8; // 16B aligned
  return cat8(*(const v8h*)p, *(const v8h*)(p + 16));
}

// C/D layout (16x16 f32, 8 VGPRs): element j -> row = j + 8*hf, col = l16.

// ========================================================================
// k0: one-time weight convert + transpose to f16 (B-operand friendly)
//     WnT[h][d] = Wn[d][h]; WmT[g][h] = Wm[h][g]; WoT[d][h] = Wo[h][d]
// ========================================================================
__global__ void k0_convert(const float* __restrict__ Wn, const float* __restrict__ Wm,
                           const float* __restrict__ Wo,
                           _Float16* __restrict__ WnT, _Float16* __restrict__ WmT,
                           _Float16* __restrict__ WoT) {
  const int tid = blockIdx.x * blockDim.x + threadIdx.x;
  if (tid < D_ * H_) {                 // Wn [D,H] -> WnT [H,D]
    const int d = tid / H_, h = tid % H_;
    WnT[(size_t)h * D_ + d] = (_Float16)Wn[tid];
  }
  if (tid < H_ * H_) {                 // Wm [H,H] -> WmT [H,H] (transposed)
    const int h = tid / H_, g = tid % H_;
    WmT[(size_t)g * H_ + h] = (_Float16)Wm[tid];
  }
  if (tid < H_ * D_) {                 // Wo [H,D] -> WoT [D,H]
    const int h = tid / D_, d = tid % D_;
    WoT[(size_t)d * H_ + h] = (_Float16)Wo[tid];
  }
}

// ========================================================================
// k1: nodes = X @ Wn + bn   -> nodes f32 + nodes f16 copy
// grid (B_*MT), block 128 (4 waves); wave w owns 16 rows x 64 cols
// ========================================================================
__global__ void k1_nodes(const float* __restrict__ X, const _Float16* __restrict__ WnT,
                         const float* __restrict__ bn,
                         float* __restrict__ nodes, _Float16* __restrict__ nodes_h) {
  const int b = blockIdx.x / MT, mt = blockIdx.x % MT;
  const int wave = threadIdx.x >> 5, lane = threadIdx.x & 31;
  const int hf = lane >> 4, l16 = lane & 15;
  const int mbase = mt * 16;
  const int hbase = wave * 64;

  const float* A = X + (size_t)b * N_ * D_;
  int rowa = mbase + l16; if (rowa >= N_) rowa = N_ - 1;

  v8f c[4] = {};
  for (int k0 = 0; k0 < D_; k0 += 32) {
    if (k0 + 32 < D_)
      __builtin_prefetch(WnT + (size_t)(hbase + l16) * D_ + k0 + 32, 0, 0);
    const v16h a = load_a_f32(A, D_, rowa, k0, hf);
#pragma unroll
    for (int t = 0; t < 4; ++t) {
      const v16h bb = load_bT_f16(WnT, D_, hbase + t * 16 + l16, k0, hf);
      c[t] = wmma_f16(a, bb, c[t]);
    }
  }

  float* Nb = nodes + (size_t)b * N_ * H_;
  _Float16* Nh = nodes_h + (size_t)b * N_ * H_;
#pragma unroll
  for (int t = 0; t < 4; ++t) {
    const int col = hbase + t * 16 + l16;
    const float bias = bn[col];
#pragma unroll
    for (int j = 0; j < 8; ++j) {
      const int grow = mbase + j + 8 * hf;
      if (grow < N_) {
        const float v = c[t][j] + bias;
        Nb[(size_t)grow * H_ + col] = v;
        Nh[(size_t)grow * H_ + col] = (_Float16)v;
      }
    }
  }
}

// ========================================================================
// k2: sim stripe (16 x 196) via WMMA (all-f16 fragments) -> LDS, mask,
//     per-row top-8, topk_vals out, neighbor gather-mean -> neighs (f16).
// grid (B_*MT), block 256 (8 waves); wave w owns n-tiles {w, w+8}
// ========================================================================
__global__ void k2_sim_topk_gather(const float* __restrict__ nodes,
                                   const _Float16* __restrict__ nodes_h,
                                   _Float16* __restrict__ neighs,
                                   float* __restrict__ topk_out) {
  __shared__ float lds_sim[16][MT * 16];   // 16 x 208 f32
  __shared__ int   lds_idx[16][KK];

  const int b = blockIdx.x / MT, mt = blockIdx.x % MT;
  const int mbase = mt * 16;
  const int wave = threadIdx.x >> 5, lane = threadIdx.x & 31;
  const int hf = lane >> 4, l16 = lane & 15;
  const float*    Nb = nodes   + (size_t)b * N_ * H_;
  const _Float16* Nh = nodes_h + (size_t)b * N_ * H_;

  int rowa = mbase + l16; if (rowa >= N_) rowa = N_ - 1;

  const int t0 = wave, t1 = wave + 8;
  const bool has2 = (t1 < MT);
  const int col0 = t0 * 16 + l16;                  // always < 196
  const int col1 = has2 ? t1 * 16 + l16 : 0;
  const int col1c = (col1 >= N_) ? N_ - 1 : col1;

  v8f c0 = {}, c1 = {};
  for (int k0 = 0; k0 < H_; k0 += 32) {
    const v16h a  = load_a_f16(Nh, H_, rowa, k0, hf);
    const v16h b0 = load_bT_f16(Nh, H_, col0, k0, hf);
    c0 = wmma_f16(a, b0, c0);
    if (has2) {                                    // wave-uniform branch
      const v16h b1 = load_bT_f16(Nh, H_, col1c, k0, hf);
      c1 = wmma_f16(a, b1, c1);
    }
  }
#pragma unroll
  for (int j = 0; j < 8; ++j) {
    const int rl = j + 8 * hf;
    const int grow = mbase + rl;
    float v = c0[j];
    if (grow == col0) v += NEGV;
    lds_sim[rl][col0] = v;
  }
  if (has2) {
#pragma unroll
    for (int j = 0; j < 8; ++j) {
      const int rl = j + 8 * hf;
      const int grow = mbase + rl;
      float v = c1[j];
      if (col1 >= N_) v = PADV;
      else if (grow == col1) v += NEGV;
      lds_sim[rl][col1] = v;
    }
  }
  __syncthreads();

  // ---- per-row top-8 (sorted desc; strict '>' keeps lowest index on ties)
  if (threadIdx.x < 16) {
    const int r = threadIdx.x;
    const int grow = mbase + r;
    if (grow < N_) {
      float tv[KK]; int ti[KK];
#pragma unroll
      for (int j = 0; j < KK; ++j) { tv[j] = PADV; ti[j] = 0; }
      for (int cc = 0; cc < N_; ++cc) {
        const float s = lds_sim[r][cc];
        if (s > tv[KK - 1]) {
          int p = KK - 1;
          while (p > 0 && tv[p - 1] < s) {
            tv[p] = tv[p - 1]; ti[p] = ti[p - 1]; --p;
          }
          tv[p] = s; ti[p] = cc;
        }
      }
#pragma unroll
      for (int j = 0; j < KK; ++j) {
        topk_out[((size_t)b * N_ + grow) * KK + j] = tv[j];
        lds_idx[r][j] = ti[j];
      }
    }
  }
  __syncthreads();

  // ---- neighbor gather + mean: thread t owns column h=t (H_ == blockDim.x)
  const int h = threadIdx.x;
  for (int r = 0; r < 16; ++r) {
    const int grow = mbase + r;
    if (grow >= N_) continue;
    float s = 0.f;
#pragma unroll
    for (int j = 0; j < KK; ++j) s += Nb[(size_t)lds_idx[r][j] * H_ + h];
    neighs[((size_t)b * N_ + grow) * H_ + h] = (_Float16)(s * (1.0f / KK));
  }
}

// ========================================================================
// k3: updated = nodes + relu(neighs @ Wm + bm), stored f16
// grid (B_*MT), block 128; wave w owns 16 rows x 64 cols
// ========================================================================
__global__ void k3_message(const _Float16* __restrict__ neighs,
                           const _Float16* __restrict__ WmT,
                           const float* __restrict__ bm,
                           const float* __restrict__ nodes,
                           _Float16* __restrict__ updated) {
  const int b = blockIdx.x / MT, mt = blockIdx.x % MT;
  const int wave = threadIdx.x >> 5, lane = threadIdx.x & 31;
  const int hf = lane >> 4, l16 = lane & 15;
  const int mbase = mt * 16;
  const int gbase = wave * 64;

  const _Float16* Ab = neighs + (size_t)b * N_ * H_;
  int rowa = mbase + l16; if (rowa >= N_) rowa = N_ - 1;

  v8f c[4] = {};
  for (int k0 = 0; k0 < H_; k0 += 32) {
    const v16h a = load_a_f16(Ab, H_, rowa, k0, hf);
#pragma unroll
    for (int t = 0; t < 4; ++t) {
      const v16h bb = load_bT_f16(WmT, H_, gbase + t * 16 + l16, k0, hf);
      c[t] = wmma_f16(a, bb, c[t]);
    }
  }

  const float* Nb = nodes + (size_t)b * N_ * H_;
  _Float16* Ub = updated + (size_t)b * N_ * H_;
#pragma unroll
  for (int t = 0; t < 4; ++t) {
    const int col = gbase + t * 16 + l16;
    const float bias = bm[col];
#pragma unroll
    for (int j = 0; j < 8; ++j) {
      const int grow = mbase + j + 8 * hf;
      if (grow < N_) {
        float m = c[t][j] + bias;
        m = m > 0.f ? m : 0.f;
        Ub[(size_t)grow * H_ + col] = (_Float16)(Nb[(size_t)grow * H_ + col] + m);
      }
    }
  }
}

// ========================================================================
// k4: out = updated @ Wo + bo  (f32 out to d_out)
// grid (B_*MT, 3), block 128; block covers 256 of 768 cols, wave 16x64
// ========================================================================
__global__ void k4_output(const _Float16* __restrict__ updated,
                          const _Float16* __restrict__ WoT,
                          const float* __restrict__ bo,
                          float* __restrict__ out) {
  const int b = blockIdx.x / MT, mt = blockIdx.x % MT;
  const int wave = threadIdx.x >> 5, lane = threadIdx.x & 31;
  const int hf = lane >> 4, l16 = lane & 15;
  const int mbase = mt * 16;
  const int dbase = blockIdx.y * 256 + wave * 64;

  const _Float16* Ab = updated + (size_t)b * N_ * H_;
  int rowa = mbase + l16; if (rowa >= N_) rowa = N_ - 1;

  v8f c[4] = {};
  for (int k0 = 0; k0 < H_; k0 += 32) {
    const v16h a = load_a_f16(Ab, H_, rowa, k0, hf);
#pragma unroll
    for (int t = 0; t < 4; ++t) {
      const v16h bb = load_bT_f16(WoT, H_, dbase + t * 16 + l16, k0, hf);
      c[t] = wmma_f16(a, bb, c[t]);
    }
  }

  const float* bop = bo;
  float* Ob = out + (size_t)b * N_ * D_;
#pragma unroll
  for (int t = 0; t < 4; ++t) {
    const int col = dbase + t * 16 + l16;
    const float bias = bop[col];
#pragma unroll
    for (int j = 0; j < 8; ++j) {
      const int grow = mbase + j + 8 * hf;
      if (grow < N_) Ob[(size_t)grow * D_ + col] = c[t][j] + bias;
    }
  }
}

// ========================================================================
extern "C" void kernel_launch(void* const* d_in, const int* in_sizes, int n_in,
                              void* d_out, int out_size, void* d_ws, size_t ws_size,
                              hipStream_t stream) {
  const float* X  = (const float*)d_in[0];   // [B,N,D]
  const float* Wn = (const float*)d_in[1];   // [D,H]
  const float* bn = (const float*)d_in[2];   // [H]
  const float* Wm = (const float*)d_in[3];   // [H,H]
  const float* bm = (const float*)d_in[4];   // [H]
  const float* Wo = (const float*)d_in[5];   // [H,D]
  const float* bo = (const float*)d_in[6];   // [D]

  float* out  = (float*)d_out;                        // [B,N,D]
  float* topk = out + (size_t)B_ * N_ * D_;           // [B,N,K]

  char* ws = (char*)d_ws;
  const size_t nodes_f32_b = (size_t)B_ * N_ * H_ * sizeof(float);     // 51.4 MB
  const size_t bnh_f16_b   = (size_t)B_ * N_ * H_ * sizeof(_Float16);  // 25.7 MB
  size_t off = 0;
  float*    nodes   = (float*)(ws + off);    off += nodes_f32_b;
  _Float16* nodes_h = (_Float16*)(ws + off); off += bnh_f16_b;
  _Float16* neighs  = (_Float16*)(ws + off); off += bnh_f16_b;
  _Float16* updated = (_Float16*)(ws + off); off += bnh_f16_b;
  _Float16* WnT     = (_Float16*)(ws + off); off += (size_t)D_ * H_ * 2;
  _Float16* WmT     = (_Float16*)(ws + off); off += (size_t)H_ * H_ * 2;
  _Float16* WoT     = (_Float16*)(ws + off); off += (size_t)H_ * D_ * 2;

  const int cvt_threads = D_ * H_;           // largest of the three
  k0_convert<<<dim3((cvt_threads + 255) / 256), dim3(256), 0, stream>>>(
      Wn, Wm, Wo, WnT, WmT, WoT);
  k1_nodes<<<dim3(B_ * MT), dim3(128), 0, stream>>>(X, WnT, bn, nodes, nodes_h);
  k2_sim_topk_gather<<<dim3(B_ * MT), dim3(256), 0, stream>>>(nodes, nodes_h,
                                                              neighs, topk);
  k3_message<<<dim3(B_ * MT), dim3(128), 0, stream>>>(neighs, WmT, bm, nodes, updated);
  k4_output<<<dim3(B_ * MT, 3), dim3(128), 0, stream>>>(updated, WoT, bo, out);
}
